// Decoder_49658411876915
// MI455X (gfx1250) — matrix operand order repaired and checked by reading, not compiled, not run on hardware
//
#include <hip/hip_runtime.h>
#include <math.h>

typedef __attribute__((ext_vector_type(16))) __bf16 v16bf;
typedef __attribute__((ext_vector_type(8)))  __bf16 v8bf;
typedef __attribute__((ext_vector_type(8)))  float  v8f;

#define ODE_THREADS 512
#define NELEM 65536            // 32*32*64 per image (HWC)
#define SM_ELEMS (34*34*32)    // padded 34x34 spatial, 32-channel half, bf16

// ---------------------------------------------------------------------------
// Async copy of one 16B chunk: global (saddr + voffset) -> LDS. ASYNCcnt path.
// ---------------------------------------------------------------------------
__device__ __forceinline__ void async_copy_b128(const __bf16* gbase, unsigned byte_off,
                                                const __bf16* ldsdst) {
    asm volatile("global_load_async_to_lds_b128 %0, %1, %2"
                 :: "v"((unsigned)(size_t)ldsdst),   // LDS byte address (addr[31:0])
                    "v"(byte_off),                   // per-lane global byte offset
                    "s"(gbase)                       // uniform 64-bit base (SGPR pair)
                 : "memory");
}
__device__ __forceinline__ void wait_async0() {
    asm volatile("s_wait_asynccnt 0" ::: "memory");
}

// ---------------------------------------------------------------------------
// Weight prep: OIHW f32 -> [layer][tap][c_out][c_in] bf16  (B-matrix friendly)
// ---------------------------------------------------------------------------
__global__ void prep_weights(const float* __restrict__ w1, const float* __restrict__ w2,
                             const float* __restrict__ w3, const float* __restrict__ w4,
                             __bf16* __restrict__ Wp) {
    int idx = blockIdx.x * blockDim.x + threadIdx.x;
    if (idx >= 4 * 9 * 64 * 64) return;
    int ci  = idx & 63;
    int r   = idx >> 6;
    int co  = r & 63;
    int r2  = r >> 6;
    int tap = r2 % 9;
    int l   = r2 / 9;
    const float* w = (l == 0) ? w1 : (l == 1) ? w2 : (l == 2) ? w3 : w4;
    Wp[idx] = (__bf16)(w[(co * 64 + ci) * 9 + tap]);   // w[co][ci][ky][kx]
}

// ---------------------------------------------------------------------------
// One 3x3 conv (64->64, pad 1) as implicit GEMM via WMMA bf16.
// src/dst: f32 HWC [32][32][64] in global (L2-hot). Wl: bf16 [9][64][64].
// A staged in LDS (padded tile); B async-double-buffered in LDS per tap.
// 16 waves, each owns 4 M-tiles x 4 N-tiles. K = 2 c-halves x 9 taps x 32.
// ---------------------------------------------------------------------------
__device__ __forceinline__ void conv3x3_wmma(
    const float* __restrict__ src, float* __restrict__ dst,
    const __bf16* __restrict__ Wl, const float* __restrict__ bias,
    int do_tanh, __bf16* sm, __bf16* wlds, int tid)
{
    const int lane = tid & 31;
    const int wv   = tid >> 5;     // wave 0..15
    const int lrow = lane & 15;
    const int side = lane >> 4;

    v8f acc[4][4];
    #pragma unroll
    for (int i = 0; i < 4; ++i)
        #pragma unroll
        for (int j = 0; j < 4; ++j)
            #pragma unroll
            for (int e = 0; e < 8; ++e) acc[i][j][e] = 0.0f;

    // Per-wave A-fragment base pointers (constant tap offsets fold into ds imm)
    const __bf16* abp[4];
    #pragma unroll
    for (int mi = 0; mi < 4; ++mi) {
        const int mt = wv * 4 + mi;
        abp[mi] = sm + (((mt >> 1) * 34) + ((mt & 1) << 4) + lrow) * 32 + side * 8;
    }

    for (int ch = 0; ch < 2; ++ch) {            // c_in halves (K split)
        __syncthreads();                        // prior smem/wlds readers done
        // Stage padded 34x34 x 32ch bf16 activation tile
        for (int i = tid; i < 34 * 34; i += ODE_THREADS) {
            const int hh = i / 34, ww = i % 34;
            __bf16* dp = sm + i * 32;
            if (hh >= 1 && hh <= 32 && ww >= 1 && ww <= 32) {
                const float* sp = src + (((hh - 1) * 32) + (ww - 1)) * 64 + ch * 32;
                #pragma unroll
                for (int c = 0; c < 32; ++c) dp[c] = (__bf16)sp[c];
            } else {
                #pragma unroll
                for (int c = 0; c < 32; ++c) dp[c] = (__bf16)0.0f;
            }
        }
        __syncthreads();
        // kick async copy of tap-0 weight block (8KB = 512 threads x 16B)
        async_copy_b128(Wl, (unsigned)(tid * 16), wlds + tid * 8);

        #pragma unroll
        for (int tap = 0; tap < 9; ++tap) {
            const int ky = tap / 3, kx = tap % 3;
            wait_async0();                      // own transfer done
            __syncthreads();                    // everyone's transfer done
            if (tap < 8)                        // overlap next tap's copy
                async_copy_b128(Wl, (unsigned)((tap + 1) * 8192 + tid * 16),
                                wlds + ((tap + 1) & 1) * 4096 + tid * 8);

            // B fragments from LDS: lane%16 = N col (co), K (ci) side*8..+7 and 16+side*8..+7
            const __bf16* bb = wlds + (tap & 1) * 4096 + lrow * 64 + ch * 32 + side * 8;
            v16bf Bf[4];
            #pragma unroll
            for (int nt = 0; nt < 4; ++nt) {
                v8bf blo = *(const v8bf*)(bb + nt * 1024);
                v8bf bhi = *(const v8bf*)(bb + nt * 1024 + 16);
                Bf[nt] = __builtin_shufflevector(blo, bhi,
                    0,1,2,3,4,5,6,7,8,9,10,11,12,13,14,15);
            }

            #pragma unroll
            for (int mi = 0; mi < 4; ++mi) {
                // A fragment: base + constant tap offset (ds offset immediate)
                const __bf16* ap = abp[mi] + (ky * 34 + kx) * 32;
                v8bf lo = *(const v8bf*)(ap);        // K side*8 .. +7
                v8bf hi = *(const v8bf*)(ap + 16);   // K 16+side*8 .. +7
                v16bf Af = __builtin_shufflevector(lo, hi,
                    0,1,2,3,4,5,6,7,8,9,10,11,12,13,14,15);
                #pragma unroll
                for (int nt = 0; nt < 4; ++nt)
                    acc[mi][nt] = __builtin_amdgcn_wmma_f32_16x16x32_bf16(
                        false, Af, false, Bf[nt], (short)0, acc[mi][nt],
                        false, false);
            }
        }
    }

    // Epilogue: D layout -> VGPR r holds (M = r + 8*side, N = lrow)
    #pragma unroll
    for (int mi = 0; mi < 4; ++mi) {
        const int mt    = wv * 4 + mi;
        const int h     = mt >> 1;
        const int wbase = (mt & 1) << 4;
        #pragma unroll
        for (int nt = 0; nt < 4; ++nt) {
            const int n  = nt * 16 + lrow;
            const float bv = bias[n];
            #pragma unroll
            for (int r = 0; r < 8; ++r) {
                const int w = wbase + r + side * 8;
                float v = acc[mi][nt][r] + bv;
                if (do_tanh) v = tanhf(v);
                dst[((h * 32 + w) * 64) + n] = v;
            }
        }
    }
    __threadfence();
    __syncthreads();
}

// ---------------------------------------------------------------------------
// Persistent ODE kernel: one block per batch image, full 18-substep RK4.
// All state buffers f32 HWC in workspace (L2-resident).
// ---------------------------------------------------------------------------
__global__ __launch_bounds__(ODE_THREADS)
void ode_kernel(const float* __restrict__ h0, const __bf16* __restrict__ Wp,
                const float* __restrict__ b1, const float* __restrict__ b2,
                const float* __restrict__ b3, const float* __restrict__ b4,
                float* __restrict__ outs, float* __restrict__ bufs)
{
    __shared__ __align__(16) __bf16 sm[SM_ELEMS];
    __shared__ __align__(16) __bf16 wlds[2 * 4096];   // per-tap weight double buffer
    const int b   = blockIdx.x;
    const int tid = threadIdx.x;

    float* Y   = bufs + (size_t)(0 * 4 + b) * NELEM;
    float* E   = bufs + (size_t)(1 * 4 + b) * NELEM;
    float* T1  = bufs + (size_t)(2 * 4 + b) * NELEM;
    float* T2  = bufs + (size_t)(3 * 4 + b) * NELEM;
    float* K   = bufs + (size_t)(4 * 4 + b) * NELEM;
    float* ACC = bufs + (size_t)(5 * 4 + b) * NELEM;

    const __bf16* W0 = Wp + 0 * 9 * 4096;
    const __bf16* W1 = Wp + 1 * 9 * 4096;
    const __bf16* W2 = Wp + 2 * 9 * 4096;
    const __bf16* W3 = Wp + 3 * 9 * 4096;

    // init: NCHW f32 -> HWC f32, also snapshot t=0
    for (int i = tid; i < NELEM; i += ODE_THREADS) {
        const int c = i & 63, s = i >> 6;
        const float v = h0[((size_t)b * 64 + c) * 1024 + s];
        Y[i] = v;
        outs[(size_t)(0 * 4 + b) * NELEM + i] = v;
    }
    __threadfence();
    __syncthreads();

    const float dt = 0.5f;

#define CHAIN(SRC)                                                \
    conv3x3_wmma(SRC, T1, W0, b1, 1, sm, wlds, tid);              \
    conv3x3_wmma(T1,  T2, W1, b2, 1, sm, wlds, tid);              \
    conv3x3_wmma(T2,  T1, W2, b3, 1, sm, wlds, tid);              \
    conv3x3_wmma(T1,  K,  W3, b4, 0, sm, wlds, tid);

    for (int step = 0; step < 18; ++step) {
        // k1
        CHAIN(Y);
        for (int i = tid; i < NELEM; i += ODE_THREADS) {
            const float k = K[i];
            ACC[i] = k;
            E[i]   = Y[i] + 0.5f * dt * k;
        }
        __threadfence(); __syncthreads();
        // k2
        CHAIN(E);
        for (int i = tid; i < NELEM; i += ODE_THREADS) {
            const float k = K[i];
            ACC[i] += 2.0f * k;
            E[i]    = Y[i] + 0.5f * dt * k;
        }
        __threadfence(); __syncthreads();
        // k3
        CHAIN(E);
        for (int i = tid; i < NELEM; i += ODE_THREADS) {
            const float k = K[i];
            ACC[i] += 2.0f * k;
            E[i]    = Y[i] + dt * k;
        }
        __threadfence(); __syncthreads();
        // k4 + combine
        CHAIN(E);
        for (int i = tid; i < NELEM; i += ODE_THREADS)
            Y[i] += (dt / 6.0f) * (ACC[i] + K[i]);
        __threadfence(); __syncthreads();

        if (step & 1) {   // snapshot after every 2 substeps (unit time)
            const int t = step / 2 + 1;
            for (int i = tid; i < NELEM; i += ODE_THREADS)
                outs[(size_t)(t * 4 + b) * NELEM + i] = Y[i];
            __syncthreads();
        }
    }
#undef CHAIN
}

// ---------------------------------------------------------------------------
// Decoder stage 1: transposed conv 64->32, k=4, s=2, p=1 + leaky relu(0.2)
// in: outs [40][32][32][64] HWC ; out: D1 [40][64][64][32] HWC
// ---------------------------------------------------------------------------
__global__ void deconv1_kernel(const float* __restrict__ outs,
                               const float* __restrict__ w,   // (32,64,4,4) OIHW
                               const float* __restrict__ bias,
                               float* __restrict__ D1)
{
    int idx = blockIdx.x * blockDim.x + threadIdx.x;
    if (idx >= 40 * 64 * 64) return;
    const int ox  = idx & 63;
    const int oy  = (idx >> 6) & 63;
    const int img = idx >> 12;
    const float* in = outs + (size_t)img * NELEM;

    float acc[32];
    #pragma unroll
    for (int oc = 0; oc < 32; ++oc) acc[oc] = bias[oc];

    for (int ky = 0; ky < 4; ++ky) {
        const int p = oy + ky - 2;
        if (p < 0 || p > 62 || (p & 1)) continue;
        const int iy = p >> 1;
        for (int kx = 0; kx < 4; ++kx) {
            const int q = ox + kx - 2;
            if (q < 0 || q > 62 || (q & 1)) continue;
            const int ix = q >> 1;
            const float* xp = in + (iy * 32 + ix) * 64;
            const float* wp = w + ky * 4 + kx;          // + ic*16 + oc*1024
            for (int ic = 0; ic < 64; ++ic) {
                const float xv = xp[ic];
                #pragma unroll
                for (int oc = 0; oc < 32; ++oc)
                    acc[oc] += xv * wp[oc * 1024 + ic * 16];
            }
        }
    }
    float* op = D1 + (size_t)idx * 32;
    #pragma unroll
    for (int oc = 0; oc < 32; ++oc) {
        const float v = acc[oc];
        op[oc] = v > 0.0f ? v : 0.2f * v;
    }
}

// ---------------------------------------------------------------------------
// Decoder stage 2: transposed conv 32->16, k=4, s=2, p=1 + leaky relu(0.2)
// in: D1 [40][64][64][32] ; out: D2 [40][128][128][16]
// ---------------------------------------------------------------------------
__global__ void deconv2_kernel(const float* __restrict__ D1,
                               const float* __restrict__ w,   // (16,32,4,4)
                               const float* __restrict__ bias,
                               float* __restrict__ D2)
{
    int idx = blockIdx.x * blockDim.x + threadIdx.x;
    if (idx >= 40 * 128 * 128) return;
    const int ox  = idx & 127;
    const int oy  = (idx >> 7) & 127;
    const int img = idx >> 14;
    const float* in = D1 + (size_t)img * 64 * 64 * 32;

    float acc[16];
    #pragma unroll
    for (int oc = 0; oc < 16; ++oc) acc[oc] = bias[oc];

    for (int ky = 0; ky < 4; ++ky) {
        const int p = oy + ky - 2;
        if (p < 0 || p > 126 || (p & 1)) continue;
        const int iy = p >> 1;
        for (int kx = 0; kx < 4; ++kx) {
            const int q = ox + kx - 2;
            if (q < 0 || q > 126 || (q & 1)) continue;
            const int ix = q >> 1;
            const float* xp = in + (iy * 64 + ix) * 32;
            const float* wp = w + ky * 4 + kx;          // + ic*16 + oc*512
            for (int ic = 0; ic < 32; ++ic) {
                const float xv = xp[ic];
                #pragma unroll
                for (int oc = 0; oc < 16; ++oc)
                    acc[oc] += xv * wp[oc * 512 + ic * 16];
            }
        }
    }
    float* op = D2 + (size_t)idx * 16;
    #pragma unroll
    for (int oc = 0; oc < 16; ++oc) {
        const float v = acc[oc];
        op[oc] = v > 0.0f ? v : 0.2f * v;
    }
}

// ---------------------------------------------------------------------------
// Decoder stage 3: 3x3 conv 16->1, pad 1. out: d_out [10][4][1][128][128]
// ---------------------------------------------------------------------------
__global__ void conv3_kernel(const float* __restrict__ D2,
                             const float* __restrict__ w,    // (1,16,3,3)
                             const float* __restrict__ bias,
                             float* __restrict__ out)
{
    int idx = blockIdx.x * blockDim.x + threadIdx.x;
    if (idx >= 40 * 128 * 128) return;
    const int ox  = idx & 127;
    const int oy  = (idx >> 7) & 127;
    const int img = idx >> 14;
    const float* in = D2 + (size_t)img * 128 * 128 * 16;

    float acc = bias[0];
    for (int ky = 0; ky < 3; ++ky) {
        const int iy = oy + ky - 1;
        if (iy < 0 || iy > 127) continue;
        for (int kx = 0; kx < 3; ++kx) {
            const int ix = ox + kx - 1;
            if (ix < 0 || ix > 127) continue;
            const float* xp = in + (iy * 128 + ix) * 16;
            const float* wp = w + ky * 3 + kx;          // + ic*9
            #pragma unroll
            for (int ic = 0; ic < 16; ++ic)
                acc += xp[ic] * wp[ic * 9];
        }
    }
    out[(size_t)img * 16384 + oy * 128 + ox] = acc;
}

// ---------------------------------------------------------------------------
extern "C" void kernel_launch(void* const* d_in, const int* in_sizes, int n_in,
                              void* d_out, int out_size, void* d_ws, size_t ws_size,
                              hipStream_t stream)
{
    const float* h0  = (const float*)d_in[0];
    const float* w1  = (const float*)d_in[1];
    const float* b1  = (const float*)d_in[2];
    const float* w2  = (const float*)d_in[3];
    const float* b2  = (const float*)d_in[4];
    const float* w3  = (const float*)d_in[5];
    const float* b3  = (const float*)d_in[6];
    const float* w4  = (const float*)d_in[7];
    const float* b4  = (const float*)d_in[8];
    const float* s1w = (const float*)d_in[9];
    const float* s1b = (const float*)d_in[10];
    const float* s2w = (const float*)d_in[11];
    const float* s2b = (const float*)d_in[12];
    const float* s3w = (const float*)d_in[13];
    const float* s3b = (const float*)d_in[14];
    float* out = (float*)d_out;

    char* ws = (char*)d_ws;
    size_t o = 0;
    float*  outs = (float*)(ws + o);  o += (size_t)10 * 4 * NELEM * 4;       // snapshots
    float*  bufs = (float*)(ws + o);  o += (size_t)6  * 4 * NELEM * 4;       // RK state
    __bf16* Wp   = (__bf16*)(ws + o); o += (size_t)4 * 9 * 64 * 64 * 2;
    o = (o + 255) & ~(size_t)255;
    float*  D1   = (float*)(ws + o);  o += (size_t)40 * 64 * 64 * 32 * 4;
    float*  D2   = (float*)(ws + o);  o += (size_t)40 * 128 * 128 * 16 * 4;

    prep_weights<<<(4 * 9 * 64 * 64 + 255) / 256, 256, 0, stream>>>(w1, w2, w3, w4, Wp);
    ode_kernel<<<4, ODE_THREADS, 0, stream>>>(h0, Wp, b1, b2, b3, b4, outs, bufs);
    deconv1_kernel<<<(40 * 64 * 64 + 255) / 256, 256, 0, stream>>>(outs, s1w, s1b, D1);
    deconv2_kernel<<<(40 * 128 * 128 + 255) / 256, 256, 0, stream>>>(D1, s2w, s2b, D2);
    conv3_kernel<<<(40 * 128 * 128 + 255) / 256, 256, 0, stream>>>(D2, s3w, s3b, out);
}
